// VQVAE_3272765079607
// MI455X (gfx1250) — compile-verified, matrix operand order checked by reading, not run on hardware
//
#include <hip/hip_runtime.h>
#include <hip/hip_bf16.h>

typedef __attribute__((ext_vector_type(16))) __bf16 v16bf;
typedef __attribute__((ext_vector_type(8)))  float  v8f;

union FR { v16bf v; uint4 q[2]; };

#define F_RELU      1
#define F_RES       2
#define F_REFLECT   4
#define F_PRELU16   8
#define F_OUT_TRANS 16
#define F_IN_TRANS  32

#define NBLK 64
#define LDS_COL 80   // bytes per staged column: 32 bf16 = 64B data + 16B pad

// ---------------------------------------------------------------------------
// Weight repack: (O, I, F) fp32 -> bf16 row-major [O][Kpad], k = tap*I + ci
// Zero-fills k >= I*F, which lets the GEMM stage B unguarded.
// ---------------------------------------------------------------------------
__global__ void pack_w(const float* __restrict__ w, __bf16* __restrict__ out,
                       int O, int I, int F, int Kpad) {
  int idx = blockIdx.x * 256 + threadIdx.x;
  if (idx >= O * Kpad) return;
  int o = idx / Kpad, k = idx - o * Kpad;
  float v = 0.f;
  if (k < I * F) {
    int tap = k / I, ci = k - tap * I;
    v = w[(size_t)(o * I + ci) * F + tap];
  }
  out[idx] = (__bf16)v;
}

// ---------------------------------------------------------------------------
// Conv-as-GEMM, bf16 WMMA 16x16x32, fp32 accumulate.
// 256 threads = 8 wave32, MW waves in M x (8/MW) in N; each wave computes
// 16M x (MW/2)*16 N with MW/2 accumulators. Block tile: (MW*16)M x 64N.
// grid = (Cout/(MW*16), Tout/64, Batch).
// Requires: Cin >= 32, nchunks = Kpad/32 even and >= 2, input tensor valid
// under 2-chunk tap overshoot (guaranteed by reflect/clamp).
// ---------------------------------------------------------------------------
template <int FLAGS, int MW>
__global__ __launch_bounds__(256)
void conv_gemm(const float* __restrict__ X, const __bf16* __restrict__ A,
               const float* __restrict__ bias, const float* __restrict__ R,
               float* __restrict__ Y,
               int Cin, int Cout, int Tin, int Tout,
               int Kpad, int stride, int padl) {
  static_assert(MW == 4 || MW == 8, "MW must be 4 or 8");
  constexpr int NS = MW / 2;        // N subtiles per wave

  __shared__ char bsm[2 * NBLK * LDS_COL];

  const int tid  = threadIdx.x;
  const int lane = tid & 31;
  const int wave = tid >> 5;
  const int wm   = wave & (MW - 1); // wave's M sub-row
  const int wn   = wave / MW;       // wave's N group
  const int b    = blockIdx.z;
  const int mbase = blockIdx.x * (MW * 16);
  const int nbase = blockIdx.y * NBLK;

  const float* Xb = X + (size_t)b * Cin * Tin;

  v8f acc[NS];
  #pragma unroll
  for (int s = 0; s < NS; ++s) acc[s] = v8f{};

  // A fragment addressing (ISA 16-bit A layout: lane m = l%16, g = l/16,
  // VGPR0-3 hold k = g*8..g*8+7, VGPR4-7 hold k = 16+g*8..16+g*8+7)
  const int am = mbase + wm * 16 + (lane & 15);
  const int ag = lane >> 4;
  const __bf16* Arow = A + (size_t)am * Kpad;

  // staging coords: sn varies fastest across threads -> coalesced X reads;
  // thread owns contiguous k = kc + skp*8 + (0..7) of each 32-k chunk.
  const int sn  = tid & 63;
  const int skp = tid >> 6;
  const int tn  = (nbase + sn) * stride - padl;

  char* buf0 = bsm;
  char* buf1 = bsm + NBLK * LDS_COL;

  // ---- staging phase 1: pure global loads into registers -----------------
  auto stage_load = [&](int ci0, int tap0, float (&f)[8]) {
    #pragma unroll
    for (int j = 0; j < 8; ++j) {
      int ci  = ci0 + j;
      int tap = tap0;
      if (ci >= Cin) { ci -= Cin; ++tap; }          // single wrap: j<8<=Cin
      int t = tn + tap;
      if (FLAGS & F_REFLECT) {                      // branchless reflect map
        t = (t < 0) ? -t : t;
        t = (t >= Tin) ? (2 * Tin - 2 - t) : t;
      } else {
        t = (t < 0) ? 0 : ((t >= Tin) ? Tin - 1 : t);
      }
      int o = (FLAGS & F_IN_TRANS) ? t * Cin + ci : ci * Tin + t;
      f[j] = Xb[o];
    }
  };

  // ---- staging phase 2: transform + fp32->bf16 + LDS store ---------------
  auto stage_store = [&](char* buf, const float (&f)[8], int ci0, int tap0) {
    unsigned* dst = (unsigned*)(buf + (size_t)sn * LDS_COL + skp * 16);
    #pragma unroll
    for (int jp = 0; jp < 4; ++jp) {
      float g[2];
      #pragma unroll
      for (int h = 0; h < 2; ++h) {
        const int j = jp * 2 + h;
        float v = f[j];
        if (!(FLAGS & F_REFLECT)) {                 // zero-pad select
          int ci  = ci0 + j;
          int tap = tap0 + ((ci >= Cin) ? 1 : 0);
          int t   = tn + tap;
          v = ((unsigned)t < (unsigned)Tin) ? v : 0.f;
        }
        if (FLAGS & F_PRELU16) v = (v > 0.f) ? 16.f * v : v;  // 4x relu(x)+x
        g[h] = v;
      }
      union { __bf16 h2[2]; unsigned u; } p;
      p.h2[0] = (__bf16)g[0];
      p.h2[1] = (__bf16)g[1];
      dst[jp] = p.u;
    }
  };

  // incremental k -> (tap, ci) load state
  int lci = skp * 8, ltap = 0;                      // skp*8 < 32 <= Cin
  float fA[8], fB[8];
  int sciA, stapA, sciB, stapB;

  // prologue: chunk 0 -> LDS buf0; chunk 1 -> fA
  stage_load(lci, ltap, fA);
  sciA = lci; stapA = ltap;
  lci += 32; if (lci >= Cin) { lci -= Cin; ++ltap; }
  stage_store(buf0, fA, sciA, stapA);
  stage_load(lci, ltap, fA);
  sciA = lci; stapA = ltap;
  lci += 32; if (lci >= Cin) { lci -= Cin; ++ltap; }

  const int nchunks = Kpad >> 5;
  const int coloff  = (wn * (NS * 16) + (lane & 15)) * LDS_COL + ag * 16;

  // One window: read bufR (chunk kc), store fst -> bufW (chunk kc+32),
  // prefetch chunk kc+64 -> fld. Branchless; tail chunks are safe garbage.
  auto window = [&](int kc, const char* bufR, char* bufW,
                    const float (&fst)[8], int sciS, int stapS,
                    float (&fld)[8], int& sciL, int& stapL) {
    __syncthreads();
    FR af, bfr[NS];
    af.q[0] = *(const uint4*)(Arow + kc + ag * 8);
    af.q[1] = *(const uint4*)(Arow + kc + 16 + ag * 8);
    #pragma unroll
    for (int s = 0; s < NS; ++s) {
      const char* cbs = bufR + coloff + s * 16 * LDS_COL;
      bfr[s].q[0] = *(const uint4*)(cbs);
      bfr[s].q[1] = *(const uint4*)(cbs + 32);
    }

    // keep the staging prefetch AFTER the fragment loads: LOADcnt retires in
    // order, so the WMMA wait then excludes the in-flight staging loads.
    __builtin_amdgcn_sched_barrier(0);

    stage_load(lci, ltap, fld);
    sciL = lci; stapL = ltap;
    lci += 32; if (lci >= Cin) { lci -= Cin; ++ltap; }

    #pragma unroll
    for (int s = 0; s < NS; ++s)
      acc[s] = __builtin_amdgcn_wmma_f32_16x16x32_bf16(
          false, af.v, false, bfr[s].v, (short)0, acc[s], false, false);

    // chunk loaded a full window ago: registers already resident, no wait
    stage_store(bufW, fst, sciS, stapS);
  };

  for (int ic = 0; ic < nchunks; ic += 2) {
    window(ic * 32,       buf0, buf1, fA, sciA, stapA, fB, sciB, stapB);
    window((ic + 1) * 32, buf1, buf0, fB, sciB, stapB, fA, sciA, stapA);
  }

  // ---- epilogue: D layout VGPR r -> M = r + (lane/16)*8, N = lane%16 ----
  const int cm0 = mbase + wm * 16 + (lane >> 4) * 8;
  #pragma unroll
  for (int s = 0; s < NS; ++s) {
    const int cn = nbase + wn * (NS * 16) + s * 16 + (lane & 15);
    #pragma unroll
    for (int r = 0; r < 8; ++r) {
      int m = cm0 + r;
      float v = acc[s][r] + bias[m];
      if (FLAGS & F_RELU) v = fmaxf(v, 0.f);
      if (FLAGS & F_RES)
        v += R[(size_t)b * Cout * Tout + (size_t)m * Tout + cn];
      if (FLAGS & F_OUT_TRANS)
        Y[(size_t)b * Tout * Cout + (size_t)cn * Cout + m] = v;
      else
        Y[(size_t)b * Cout * Tout + (size_t)m * Tout + cn] = v;
    }
  }
}

// ---------------------------------------------------------------------------
// Vector quantization: one block per token n; 256 threads = 256 codes.
// ---------------------------------------------------------------------------
__global__ __launch_bounds__(256)
void vq_kernel(const float* __restrict__ z, const float* __restrict__ cb,
               float* __restrict__ quant, float* __restrict__ partial,
               unsigned* __restrict__ counts) {
  __shared__ float sz[64];
  __shared__ float sdist[256];
  __shared__ int   sidx[256];
  const int n = blockIdx.x;
  const int tid = threadIdx.x;
  if (tid < 64) sz[tid] = z[(size_t)n * 64 + tid];
  __syncthreads();

  const float* c = cb + (size_t)tid * 64;
  float d = 0.f;
  #pragma unroll 8
  for (int i = 0; i < 64; ++i) {
    float df = sz[i] - c[i];
    d = fmaf(df, df, d);
  }
  sdist[tid] = d;
  sidx[tid]  = tid;
  __syncthreads();
  for (int s = 128; s > 0; s >>= 1) {
    if (tid < s) {
      float od = sdist[tid + s]; int oi = sidx[tid + s];
      if (od < sdist[tid] || (od == sdist[tid] && oi < sidx[tid])) {
        sdist[tid] = od; sidx[tid] = oi;     // first-min tie-break == argmin
      }
    }
    __syncthreads();
  }
  const int idx = sidx[0];

  float sq = 0.f;
  if (tid < 64) {
    float e = cb[(size_t)idx * 64 + tid];
    quant[(size_t)n * 64 + tid] = e;
    float df = sz[tid] - e;
    sq = df * df;
  }
  sdist[tid] = sq;
  __syncthreads();
  for (int s = 32; s > 0; s >>= 1) {
    if (tid < s) sdist[tid] += sdist[tid + s];
    __syncthreads();
  }
  if (tid == 0) {
    partial[n] = sdist[0];
    atomicAdd(&counts[idx], 1u);
  }
}

__global__ void zero_counts(unsigned* counts) { counts[threadIdx.x] = 0u; }

__global__ __launch_bounds__(256)
void vq_finalize(const float* __restrict__ partial,
                 const unsigned* __restrict__ counts,
                 float* __restrict__ out_loss, float* __restrict__ out_perp,
                 int N) {
  __shared__ float s1[256], s2[256];
  const int tid = threadIdx.x;
  float ls = 0.f;
  for (int i = tid; i < N; i += 256) ls += partial[i];
  float p = (float)counts[tid] / (float)N;
  s1[tid] = ls;
  s2[tid] = p * logf(p + 1e-10f);
  __syncthreads();
  for (int s = 128; s > 0; s >>= 1) {
    if (tid < s) { s1[tid] += s1[tid + s]; s2[tid] += s2[tid + s]; }
    __syncthreads();
  }
  if (tid == 0) {
    *out_loss = 1.25f * s1[0] / (float)(N * 64);   // (1+beta)*mean((z-e)^2)
    *out_perp = expf(-s2[0]);
  }
}

// ---------------------------------------------------------------------------
// Host-side orchestration
// ---------------------------------------------------------------------------
extern "C" void kernel_launch(void* const* d_in, const int* in_sizes, int n_in,
                              void* d_out, int out_size, void* d_ws, size_t ws_size,
                              hipStream_t stream) {
  (void)in_sizes; (void)n_in; (void)ws_size;
  const int B = 16, HID = 768, D = 64, T0 = 4096;

  const float* c = (const float*)d_in[0];
  const float* cw[6], *cb[6];
  for (int i = 0; i < 6; ++i) { cw[i] = (const float*)d_in[1 + 2 * i];
                                cb[i] = (const float*)d_in[2 + 2 * i]; }
  const float* lin_w    = (const float*)d_in[13];
  const float* lin_b    = (const float*)d_in[14];
  const float* codebook = (const float*)d_in[15];
  const float* pconv_w  = (const float*)d_in[16];
  const float* pconv_b  = (const float*)d_in[17];
  float* out = (float*)d_out;

  // ---- workspace carve-up (256B aligned) ----
  char* ws = (char*)d_ws;
  size_t off = 0;
  auto take = [&](size_t bytes) { char* p = ws + off;
                                  off += (bytes + 255) & ~(size_t)255; return p; };
  float*    xA    = (float*)take((size_t)B * HID * T0 * 4);
  float*    xB    = (float*)take((size_t)B * HID * T0 * 4);
  float*    zbuf  = (float*)take((size_t)B * 1024 * D * 4);   // (B,T,D)
  float*    quant = (float*)take((size_t)B * 1024 * D * 4);   // (B,T,D)
  __bf16*   wp0   = (__bf16*)take((size_t)HID * 128  * 2);
  __bf16*   wp1   = (__bf16*)take((size_t)HID * 2304 * 2);
  __bf16*   wp2   = (__bf16*)take((size_t)HID * 3072 * 2);
  __bf16*   wp3   = (__bf16*)take((size_t)HID * 3072 * 2);
  __bf16*   wp4   = (__bf16*)take((size_t)HID * 2304 * 2);
  __bf16*   wp5   = (__bf16*)take((size_t)HID * 2304 * 2);
  __bf16*   wlin  = (__bf16*)take((size_t)D * HID * 2);
  __bf16*   wpc   = (__bf16*)take((size_t)128 * 192 * 2);
  float*    part  = (float*)take((size_t)B * 1024 * 4);
  unsigned* cnts  = (unsigned*)take(256 * 4);

  auto packgrid = [](int O, int Kpad) { return dim3((O * Kpad + 255) / 256); };

  // ---- weight repack (fp32 -> bf16 GEMM layout, zero-padded K) ----
  pack_w<<<packgrid(HID, 128 ), 256, 0, stream>>>(cw[0], wp0, HID, 39,  3, 128);
  pack_w<<<packgrid(HID, 2304), 256, 0, stream>>>(cw[1], wp1, HID, HID, 3, 2304);
  pack_w<<<packgrid(HID, 3072), 256, 0, stream>>>(cw[2], wp2, HID, HID, 4, 3072);
  pack_w<<<packgrid(HID, 3072), 256, 0, stream>>>(cw[3], wp3, HID, HID, 4, 3072);
  pack_w<<<packgrid(HID, 2304), 256, 0, stream>>>(cw[4], wp4, HID, HID, 3, 2304);
  pack_w<<<packgrid(HID, 2304), 256, 0, stream>>>(cw[5], wp5, HID, HID, 3, 2304);
  pack_w<<<packgrid(D,   768 ), 256, 0, stream>>>(lin_w,   wlin, D,   HID, 1, 768);
  pack_w<<<packgrid(128, 192 ), 256, 0, stream>>>(pconv_w, wpc,  128, D,   3, 192);

  constexpr int FC  = F_RELU | F_REFLECT;           // plain conv
  constexpr int FCR = F_RELU | F_REFLECT | F_RES;   // conv + residual
  constexpr int FL  = F_PRELU16 | F_OUT_TRANS;      // linear (+4x relu-res)
  constexpr int FP  = F_IN_TRANS;                   // post conv (zero pad)

  // ---- encoder conv stack (M_BLK=128: grid.x = 768/128 = 6) ----
  conv_gemm<FC , 8><<<dim3(6, 64, B), 256, 0, stream>>>(c,  wp0, cb[0], nullptr, xA,
      39,  HID, 4096, 4096, 128, 1, 1);
  conv_gemm<FCR, 8><<<dim3(6, 64, B), 256, 0, stream>>>(xA, wp1, cb[1], xA, xB,
      HID, HID, 4096, 4096, 2304, 1, 1);
  conv_gemm<FC , 8><<<dim3(6, 32, B), 256, 0, stream>>>(xB, wp2, cb[2], nullptr, xA,
      HID, HID, 4096, 2048, 3072, 2, 2);
  conv_gemm<FC , 8><<<dim3(6, 16, B), 256, 0, stream>>>(xA, wp3, cb[3], nullptr, xB,
      HID, HID, 2048, 1024, 3072, 2, 2);
  conv_gemm<FCR, 8><<<dim3(6, 16, B), 256, 0, stream>>>(xB, wp4, cb[4], xB, xA,
      HID, HID, 1024, 1024, 2304, 1, 1);
  conv_gemm<FCR, 8><<<dim3(6, 16, B), 256, 0, stream>>>(xA, wp5, cb[5], xA, xB,
      HID, HID, 1024, 1024, 2304, 1, 1);

  // ---- linear (768 -> 64, M_BLK=64) with 4x relu(x)+x fused ----
  conv_gemm<FL , 4><<<dim3(1, 16, B), 256, 0, stream>>>(xB, wlin, lin_b, nullptr, zbuf,
      HID, D, 1024, 1024, 768, 1, 0);

  // ---- vector quantization ----
  zero_counts<<<1, 256, 0, stream>>>(cnts);
  vq_kernel<<<B * 1024, 256, 0, stream>>>(zbuf, codebook, quant, part, cnts);

  // ---- post conv (zero padding, M_BLK=128), straight into d_out ----
  conv_gemm<FP , 8><<<dim3(1, 16, B), 256, 0, stream>>>(quant, wpc, pconv_b, nullptr, out,
      D, 128, 1024, 1024, 192, 1, 1);

  // ---- scalars: d_out[out_size-2]=vq_loss, d_out[out_size-1]=perp ----
  vq_finalize<<<1, 256, 0, stream>>>(part, cnts,
      out + (out_size - 2), out + (out_size - 1), B * 1024);
}
